// SPv6_61701500174672
// MI455X (gfx1250) — matrix operand order repaired and checked by reading, not compiled, not run on hardware
//
#include <hip/hip_runtime.h>
#include <hip/hip_bf16.h>
#include <math.h>

typedef __attribute__((ext_vector_type(16))) _Float16 v16h;
typedef __attribute__((ext_vector_type(8)))  _Float16 v8h;
typedef __attribute__((ext_vector_type(8)))  float    v8f;
typedef __attribute__((ext_vector_type(4)))  unsigned int v4u;
typedef __attribute__((ext_vector_type(8)))  unsigned int v8u;

#define VOX   262144      // 64*64*64
#define CH_T  4194304     // 16*VOX (one 16-channel tensor)

// workspace float offsets
#define OFF_HX   0
#define OFF_Q    524288
#define OFF_K    532480
#define OFF_V    540672
#define OFF_XYZ  548864
#define OFF_SP1  557056
#define OFF_SP3  1081344
#define OFF_SP5A 1605632
#define OFF_SP5  2129920
#define OFF_HX2  2654208
#define OFF_Z1   3178496
#define OFF_Z2   7372800

__device__ __forceinline__ float relu_f(float x) { return fmaxf(x, 0.f); }

__device__ __forceinline__ v16h cat16(v8h lo, v8h hi) {
    return __builtin_shufflevector(lo, hi, 0, 1, 2, 3, 4, 5, 6, 7,
                                           8, 9, 10, 11, 12, 13, 14, 15);
}
// A-fragment: two contiguous 8-half runs at p and p+16 (row-major A)
__device__ __forceinline__ v16h ldfragA(const _Float16* p) {
    v8h lo = *(const v8h*)p;
    v8h hi = *(const v8h*)(p + 16);
    return cat16(lo, hi);
}
// B-fragment: one contiguous 16-half run at p (B stored transposed: [n][k])
__device__ __forceinline__ v16h ldfragB(const _Float16* p) {
    v8h lo = *(const v8h*)p;
    v8h hi = *(const v8h*)(p + 8);
    return cat16(lo, hi);
}

// ---------------------------------------------------------------------------
// K1: hx = relu(conv3d(concat(x1,x2), w_red[1,32,3,3,3], pad1) + b_red)
// ---------------------------------------------------------------------------
__global__ void k_conv_red(const float* __restrict__ x1, const float* __restrict__ x2,
                           const float* __restrict__ w, const float* __restrict__ bias,
                           float* __restrict__ hx) {
    __shared__ float wl[32 * 27];
    __shared__ float bl;
    int tid = threadIdx.x;
    for (int i = tid; i < 32 * 27; i += 256) wl[i] = w[i];
    if (tid == 0) bl = bias[0];
    __syncthreads();

    int idx = blockIdx.x * 256 + tid;            // [0, 524288)
    int b = idx >> 18;
    int z = (idx >> 12) & 63, y = (idx >> 6) & 63, x = idx & 63;
    float acc = bl;
    for (int t = 0; t < 27; ++t) {
        int kz = t / 9, ky = (t % 9) / 3, kx = t % 3;
        int gz = z + kz - 1, gy = y + ky - 1, gx = x + kx - 1;
        if ((unsigned)gz < 64u && (unsigned)gy < 64u && (unsigned)gx < 64u) {
            int off = gz * 4096 + gy * 64 + gx;
            #pragma unroll 4
            for (int c = 0; c < 32; ++c) {
                const float* src = (c < 16) ? (x1 + (b * 16 + c) * VOX)
                                            : (x2 + (b * 16 + (c - 16)) * VOX);
                acc += wl[c * 27 + t] * src[off];
            }
        }
    }
    hx[idx] = relu_f(acc);
}

// ---------------------------------------------------------------------------
// K2: q/k/v axis-collapsing convs
// ---------------------------------------------------------------------------
__global__ void k_qkv(const float* __restrict__ hx,
                      const float* __restrict__ wcx, const float* __restrict__ bcx,
                      const float* __restrict__ wcy, const float* __restrict__ bcy,
                      const float* __restrict__ wcz, const float* __restrict__ bcz,
                      float* __restrict__ q, float* __restrict__ kk, float* __restrict__ v) {
    int idx = blockIdx.x * 256 + threadIdx.x;    // [0, 3*8192)
    int which = idx >> 13;
    int r = idx & 8191;
    int b = r >> 12, s = r & 4095;
    int a0 = s >> 6, a1 = s & 63;
    const float* base = hx + b * VOX;
    float acc;
    if (which == 0) {
        acc = bcx[0];
        const float* p = base + (a0 * 64 + a1) * 64;
        for (int d = 0; d < 64; ++d) acc += p[d] * wcx[d];
        q[r] = relu_f(acc);
    } else if (which == 1) {
        acc = bcy[0];
        const float* p = base + a0 * 4096 + a1;
        for (int w = 0; w < 64; ++w) acc += p[w * 64] * wcy[w];
        kk[r] = relu_f(acc);
    } else {
        acc = bcz[0];
        const float* p = base + a1 * 64 + a0;
        for (int h = 0; h < 64; ++h) acc += p[h * 4096] * wcz[h];
        v[r] = relu_f(acc);
    }
}

// ---------------------------------------------------------------------------
// K3: xyz[s] = softmax_t(Q[s]*K[t]) @ V  -- streaming, rank-1 scores
// ---------------------------------------------------------------------------
__global__ void k_axis_attn(const float* __restrict__ q, const float* __restrict__ k,
                            const float* __restrict__ v, float* __restrict__ xyz) {
    __shared__ float lk[4096];
    __shared__ float lv[4096];
    __shared__ float red[256];
    __shared__ float smax, smin;
    int tid = threadIdx.x;
    int b = blockIdx.x >> 4;
    int rowbase = (blockIdx.x & 15) * 256;
    for (int i = tid; i < 4096; i += 256) { lk[i] = k[b * 4096 + i]; lv[i] = v[b * 4096 + i]; }
    __syncthreads();
    float mx = -1e30f;
    for (int i = tid; i < 4096; i += 256) mx = fmaxf(mx, lk[i]);
    red[tid] = mx; __syncthreads();
    for (int s = 128; s > 0; s >>= 1) { if (tid < s) red[tid] = fmaxf(red[tid], red[tid + s]); __syncthreads(); }
    if (tid == 0) smax = red[0];
    __syncthreads();
    float mn = 1e30f;
    for (int i = tid; i < 4096; i += 256) mn = fminf(mn, lk[i]);
    red[tid] = mn; __syncthreads();
    for (int s = 128; s > 0; s >>= 1) { if (tid < s) red[tid] = fminf(red[tid], red[tid + s]); __syncthreads(); }
    if (tid == 0) smin = red[0];
    __syncthreads();

    int s = rowbase + tid;
    float Q = q[b * 4096 + s];
    float m = (Q >= 0.f) ? Q * smax : Q * smin;
    float sum = 0.f, sv = 0.f;
    for (int t = 0; t < 4096; ++t) {
        float e = expf(Q * lk[t] - m);
        sum += e; sv += e * lv[t];
    }
    xyz[b * 4096 + s] = sv / sum;
}

// ---------------------------------------------------------------------------
// K4: sp1 (1x1x1), sp3 (3x3x3), sp5a (3x3x3) from hx
// ---------------------------------------------------------------------------
__global__ void k_sp(const float* __restrict__ hx,
                     const float* __restrict__ w1, const float* __restrict__ b1,
                     const float* __restrict__ w3, const float* __restrict__ b3,
                     const float* __restrict__ w5a, const float* __restrict__ b5a,
                     float* __restrict__ sp1, float* __restrict__ sp3, float* __restrict__ sp5a) {
    __shared__ float wl3[27], wl5[27], sc[4];
    int tid = threadIdx.x;
    if (tid < 27) { wl3[tid] = w3[tid]; wl5[tid] = w5a[tid]; }
    if (tid == 0) { sc[0] = w1[0]; sc[1] = b1[0]; sc[2] = b3[0]; sc[3] = b5a[0]; }
    __syncthreads();
    int idx = blockIdx.x * 256 + tid;
    int b = idx >> 18;
    int z = (idx >> 12) & 63, y = (idx >> 6) & 63, x = idx & 63;
    const float* base = hx + b * VOX;
    float a3 = sc[2], a5 = sc[3];
    for (int t = 0; t < 27; ++t) {
        int kz = t / 9, ky = (t % 9) / 3, kx = t % 3;
        int gz = z + kz - 1, gy = y + ky - 1, gx = x + kx - 1;
        float tv = 0.f;
        if ((unsigned)gz < 64u && (unsigned)gy < 64u && (unsigned)gx < 64u)
            tv = base[gz * 4096 + gy * 64 + gx];
        a3 += wl3[t] * tv;
        a5 += wl5[t] * tv;
    }
    float c = hx[idx];
    sp1[idx]  = relu_f(sc[0] * c + sc[1]);
    sp3[idx]  = relu_f(a3);
    sp5a[idx] = relu_f(a5);
}

// K5: sp5 = relu(conv3x3x3(sp5a) + b)
__global__ void k_sp5b(const float* __restrict__ sp5a, const float* __restrict__ w,
                       const float* __restrict__ bias, float* __restrict__ sp5) {
    __shared__ float wl[27], bl;
    int tid = threadIdx.x;
    if (tid < 27) wl[tid] = w[tid];
    if (tid == 0) bl = bias[0];
    __syncthreads();
    int idx = blockIdx.x * 256 + tid;
    int b = idx >> 18;
    int z = (idx >> 12) & 63, y = (idx >> 6) & 63, x = idx & 63;
    const float* base = sp5a + b * VOX;
    float acc = bl;
    for (int t = 0; t < 27; ++t) {
        int kz = t / 9, ky = (t % 9) / 3, kx = t % 3;
        int gz = z + kz - 1, gy = y + ky - 1, gx = x + kx - 1;
        if ((unsigned)gz < 64u && (unsigned)gy < 64u && (unsigned)gx < 64u)
            acc += wl[t] * base[gz * 4096 + gy * 64 + gx];
    }
    sp5[idx] = relu_f(acc);
}

// ---------------------------------------------------------------------------
// K6 (WMMA): per (b,h): attn = softmax_rows(sp1@sp3) @ sp5,
//            hx2 = xyz(broadcast) + attn + hx.  128 threads = 4 waves.
// B operands staged TRANSPOSED so every fragment is b128 LDS traffic.
// ---------------------------------------------------------------------------
__global__ void k_attn_wmma(const float* __restrict__ sp1, const float* __restrict__ sp3,
                            const float* __restrict__ sp5, const float* __restrict__ xyz,
                            const float* __restrict__ hx, float* __restrict__ hx2) {
    __shared__ __align__(16) _Float16 lA[4096];     // sp1 slice, row-major
    __shared__ __align__(16) _Float16 lBt[4096];    // sp3 slice, transposed [n][k]
    __shared__ __align__(16) _Float16 lB2t[4096];   // sp5 slice, transposed [n][k]
    __shared__ __align__(16) float    lM[4][1024];
    __shared__ __align__(16) _Float16 lP[4][1024];

    int tid = threadIdx.x;
    int bh  = blockIdx.x;                 // b*64 + h
    int off = bh * 4096;
    for (int i = tid; i < 4096; i += 128) {
        int kr = i >> 6, jc = i & 63;
        lA[i]             = (_Float16)sp1[off + i];
        lBt[jc * 64 + kr] = (_Float16)sp3[off + i];
        lB2t[jc * 64 + kr]= (_Float16)sp5[off + i];
    }
    __syncthreads();

    int wv = tid >> 5, lane = tid & 31;
    int half = lane >> 4, mn = lane & 15;
    int r0 = wv * 16;

    v8f acc[4] = {v8f{}, v8f{}, v8f{}, v8f{}};
    #pragma unroll
    for (int kc = 0; kc < 2; ++kc) {
        v16h a = ldfragA(&lA[(r0 + mn) * 64 + 32 * kc + 8 * half]);
        #pragma unroll
        for (int nt = 0; nt < 4; ++nt) {
            v16h bf = ldfragA(&lBt[(16 * nt + mn) * 64 + 32 * kc + 8 * half]);
            acc[nt] = __builtin_amdgcn_wmma_f32_16x16x32_f16(
                false, a, false, bf, (short)0, acc[nt], false, false);
        }
    }
    #pragma unroll
    for (int nt = 0; nt < 4; ++nt)
        #pragma unroll
        for (int r = 0; r < 8; ++r)
            lM[wv][(r + 8 * half) * 64 + 16 * nt + mn] = acc[nt][r];

    if (lane < 16) {
        int base = lane * 64;
        float mx = -1e30f;
        for (int j = 0; j < 64; ++j) mx = fmaxf(mx, lM[wv][base + j]);
        float sm = 0.f;
        for (int j = 0; j < 64; ++j) sm += expf(lM[wv][base + j] - mx);
        float inv = 1.f / sm;
        for (int j = 0; j < 64; ++j)
            lP[wv][base + j] = (_Float16)(expf(lM[wv][base + j] - mx) * inv);
    }

    v8f acc2[4] = {v8f{}, v8f{}, v8f{}, v8f{}};
    #pragma unroll
    for (int kc = 0; kc < 2; ++kc) {
        v16h a = ldfragA(&lP[wv][mn * 64 + 32 * kc + 8 * half]);
        #pragma unroll
        for (int nt = 0; nt < 4; ++nt) {
            v16h bf = ldfragA(&lB2t[(16 * nt + mn) * 64 + 32 * kc + 8 * half]);
            acc2[nt] = __builtin_amdgcn_wmma_f32_16x16x32_f16(
                false, a, false, bf, (short)0, acc2[nt], false, false);
        }
    }

    int b = bh >> 6, h = bh & 63;
    #pragma unroll
    for (int nt = 0; nt < 4; ++nt) {
        int jcol = 16 * nt + mn;
        #pragma unroll
        for (int r = 0; r < 8; ++r) {
            int i = r0 + r + 8 * half;
            float val = acc2[nt][r] + xyz[b * 4096 + h * 64 + i] + hx[off + i * 64 + jcol];
            hx2[off + i * 64 + jcol] = val;
        }
    }
}

// ---------------------------------------------------------------------------
// K7: sp = sigmoid(relu(conv_fus(hx2)+b) + x); z = sp*x + 2x
// ---------------------------------------------------------------------------
__global__ void k_fuse(const float* __restrict__ hx2,
                       const float* __restrict__ x1, const float* __restrict__ x2,
                       const float* __restrict__ wf, const float* __restrict__ bf,
                       float* __restrict__ z1, float* __restrict__ z2) {
    __shared__ float wl[32 * 27];
    __shared__ float bl[32];
    int tid = threadIdx.x;
    for (int i = tid; i < 32 * 27; i += 256) wl[i] = wf[i];
    if (tid < 32) bl[tid] = bf[tid];
    __syncthreads();

    int idx = blockIdx.x * 256 + tid;
    int b = idx >> 18;
    int vox = idx & (VOX - 1);
    int z = (idx >> 12) & 63, y = (idx >> 6) & 63, x = idx & 63;
    const float* base = hx2 + b * VOX;
    float tap[27];
    for (int t = 0; t < 27; ++t) {
        int kz = t / 9, ky = (t % 9) / 3, kx = t % 3;
        int gz = z + kz - 1, gy = y + ky - 1, gx = x + kx - 1;
        tap[t] = ((unsigned)gz < 64u && (unsigned)gy < 64u && (unsigned)gx < 64u)
                     ? base[gz * 4096 + gy * 64 + gx] : 0.f;
    }
    for (int c = 0; c < 32; ++c) {
        float conv = bl[c];
        #pragma unroll
        for (int t = 0; t < 27; ++t) conv += wl[c * 27 + t] * tap[t];
        float cv = relu_f(conv);
        const float* xin = (c < 16) ? x1 : x2;
        float xv = xin[(b * 16 + (c & 15)) * VOX + vox];
        float sg = 1.f / (1.f + expf(-(cv + xv)));
        float zz = sg * xv + 2.f * xv;
        float* zout = (c < 16) ? z1 : z2;
        zout[(b * 16 + (c & 15)) * VOX + vox] = zz;
    }
}

// ---------------------------------------------------------------------------
// K8 (WMMA implicit GEMM): y = relu(conv3d(z, w[16,16,3,3,3], pad1) + b)
// block (4 waves) per (which,b,z,y). Weights fetched via the Tensor Data
// Mover into LDS staging; input tile re-laid out [kz][ky][x][ic] and weights
// [n][tap][ic] so every WMMA fragment is contiguous b128 LDS traffic.
// ---------------------------------------------------------------------------
__global__ void k_conv_c(const float* __restrict__ z1, const float* __restrict__ z2,
                         const float* __restrict__ wc1, const float* __restrict__ bc1,
                         const float* __restrict__ wc2, const float* __restrict__ bc2,
                         float* __restrict__ out) {
    __shared__ __align__(16) _Float16 zin[3 * 3 * 66 * 16];  // [kz][ky][x+1][ic]
    __shared__ __align__(16) _Float16 wl[16 * 27 * 16];      // [n][tap][ic]
    __shared__ __align__(16) float    wstage[16 * 16 * 27];  // raw TDM-loaded weights
    __shared__ float bl[16];

    int tid = threadIdx.x;
    int bid = blockIdx.x;
    int y  = bid & 63;
    int zc = (bid >> 6) & 63;
    int b  = (bid >> 12) & 1;
    int which = bid >> 13;

    const float* src  = which ? z2 : z1;
    const float* wsrc = which ? wc2 : wc1;
    const float* bsrc = which ? bc2 : bc1;

    // ---- TDM: DMA the 16*16*27 f32 weight block into LDS staging (wave 0) ----
    if (tid < 32) {
        unsigned ldsoff = (unsigned)(size_t)(&wstage[0]);
        unsigned long long ga = (unsigned long long)(size_t)wsrc;
        v4u g0;
        g0[0] = 1u;                                         // count=1, user mode
        g0[1] = ldsoff;                                     // lds_addr
        g0[2] = (unsigned)(ga & 0xffffffffu);               // global_addr[31:0]
        g0[3] = (unsigned)((ga >> 32) & 0x01ffffffu)        // global_addr[56:32]
                | 0x80000000u;                              // type=2 ("image")
        v8u g1;
        g1[0] = 0x00020000u;                                // data_size=4B
        g1[1] = (6912u & 0xffffu) << 16;                    // tensor_dim0[15:0]
        g1[2] = (6912u >> 16) | (1u << 16);                 // tensor_dim0[31:16] | tensor_dim1 lo
        g1[3] = (6912u << 16);                              // tile_dim0 (bits 127:112)
        g1[4] = 0u;                                         // tile_dim1=0, tile_dim2=0
        g1[5] = 6912u;                                      // tensor_dim0_stride[31:0]
        g1[6] = (6912u & 0xffffu) << 16;                    // dim0_str hi=0 | dim1_str lo
        g1[7] = (6912u >> 16);                              // dim1_stride hi
        asm volatile("tensor_load_to_lds %0, %1" :: "s"(g0), "s"(g1) : "memory");
        __builtin_amdgcn_s_wait_tensorcnt(0);
    }

    // ---- input halo tile: [kz][ky][xx 0..65][ic] as f16 ----
    for (int i = tid; i < 3 * 3 * 66 * 16; i += 128) {
        int ic = i & 15;
        int p  = i >> 4;            // 0..593
        int xx = p % 66;
        int qq = p / 66;
        int yy = qq % 3, zz = qq / 3;
        int gz = zc + zz - 1, gy = y + yy - 1, gx = xx - 1;
        float v = 0.f;
        if ((unsigned)gz < 64u && (unsigned)gy < 64u && (unsigned)gx < 64u)
            v = src[(b * 16 + ic) * VOX + gz * 4096 + gy * 64 + gx];
        zin[i] = (_Float16)v;
    }
    if (tid < 16) bl[tid] = bsrc[tid];
    __syncthreads();

    // ---- re-layout weights: [n][ic][tap] (f32 staging) -> [n][tap][ic] f16 ----
    for (int i = tid; i < 6912; i += 128) {
        int n = i / 432;
        int r = i % 432;
        int ic = r / 27, tap = r % 27;
        wl[(n * 27 + tap) * 16 + ic] = (_Float16)wstage[i];
    }
    __syncthreads();

    int wv = tid >> 5, lane = tid & 31;
    int half = lane >> 4, mn = lane & 15;
    int x0 = wv * 16;

    v8h hz = {};
    v8f acc = v8f{};
    #pragma unroll
    for (int tp = 0; tp < 14; ++tp) {
        int t0 = 2 * tp, t1 = t0 + 1;
        // A fragment: elements 0-7 from tap t0, 8-15 from tap t1 (ic runs)
        int kz0 = t0 / 9, ky0 = (t0 % 9) / 3, kx0 = t0 % 3;
        v8h lo = *(const v8h*)&zin[(((kz0 * 3 + ky0) * 66) + x0 + mn + kx0) * 16 + 8 * half];
        v8h hi = hz;
        if (t1 < 27) {
            int kz1 = t1 / 9, ky1 = (t1 % 9) / 3, kx1 = t1 % 3;
            hi = *(const v8h*)&zin[(((kz1 * 3 + ky1) * 66) + x0 + mn + kx1) * 16 + 8 * half];
        }
        v16h a = cat16(lo, hi);
        // B fragment: elements 0-7 from tap t0, 8-15 from tap t1 (ic runs 8*half..+7)
        v8h blo = *(const v8h*)&wl[(mn * 27 + t0) * 16 + 8 * half];
        v8h bhi = hz;
        if (t1 < 27) bhi = *(const v8h*)&wl[(mn * 27 + t1) * 16 + 8 * half];
        v16h bf = cat16(blo, bhi);
        acc = __builtin_amdgcn_wmma_f32_16x16x32_f16(
            false, a, false, bf, (short)0, acc, false, false);
    }

    #pragma unroll
    for (int r = 0; r < 8; ++r) {
        int xx = x0 + r + 8 * half;
        float val = relu_f(acc[r] + bl[mn]);
        out[which * CH_T + (b * 16 + mn) * VOX + zc * 4096 + y * 64 + xx] = val;
    }
}

// ---------------------------------------------------------------------------
extern "C" void kernel_launch(void* const* d_in, const int* in_sizes, int n_in,
                              void* d_out, int out_size, void* d_ws, size_t ws_size,
                              hipStream_t stream) {
    const float* x1    = (const float*)d_in[0];
    const float* x2    = (const float*)d_in[1];
    const float* w_red = (const float*)d_in[2];
    const float* b_red = (const float*)d_in[3];
    const float* w_cx  = (const float*)d_in[4];
    const float* b_cx  = (const float*)d_in[5];
    const float* w_cy  = (const float*)d_in[6];
    const float* b_cy  = (const float*)d_in[7];
    const float* w_cz  = (const float*)d_in[8];
    const float* b_cz  = (const float*)d_in[9];
    const float* w_sp1 = (const float*)d_in[10];
    const float* b_sp1 = (const float*)d_in[11];
    const float* w_sp3 = (const float*)d_in[12];
    const float* b_sp3 = (const float*)d_in[13];
    const float* w_5a  = (const float*)d_in[14];
    const float* b_5a  = (const float*)d_in[15];
    const float* w_5b  = (const float*)d_in[16];
    const float* b_5b  = (const float*)d_in[17];
    const float* w_fus = (const float*)d_in[18];
    const float* b_fus = (const float*)d_in[19];
    const float* w_c1  = (const float*)d_in[20];
    const float* b_c1  = (const float*)d_in[21];
    const float* w_c2  = (const float*)d_in[22];
    const float* b_c2  = (const float*)d_in[23];

    float* ws  = (float*)d_ws;
    float* hx  = ws + OFF_HX;
    float* q   = ws + OFF_Q;
    float* k   = ws + OFF_K;
    float* v   = ws + OFF_V;
    float* xyz = ws + OFF_XYZ;
    float* sp1 = ws + OFF_SP1;
    float* sp3 = ws + OFF_SP3;
    float* s5a = ws + OFF_SP5A;
    float* sp5 = ws + OFF_SP5;
    float* hx2 = ws + OFF_HX2;
    float* z1  = ws + OFF_Z1;
    float* z2  = ws + OFF_Z2;
    float* out = (float*)d_out;

    k_conv_red<<<2048, 256, 0, stream>>>(x1, x2, w_red, b_red, hx);
    k_qkv<<<96, 256, 0, stream>>>(hx, w_cx, b_cx, w_cy, b_cy, w_cz, b_cz, q, k, v);
    k_axis_attn<<<32, 256, 0, stream>>>(q, k, v, xyz);
    k_sp<<<2048, 256, 0, stream>>>(hx, w_sp1, b_sp1, w_sp3, b_sp3, w_5a, b_5a, sp1, sp3, s5a);
    k_sp5b<<<2048, 256, 0, stream>>>(s5a, w_5b, b_5b, sp5);
    k_attn_wmma<<<128, 128, 0, stream>>>(sp1, sp3, sp5, xyz, hx, hx2);
    k_fuse<<<2048, 256, 0, stream>>>(hx2, x1, x2, w_fus, b_fus, z1, z2);
    k_conv_c<<<16384, 128, 0, stream>>>(z1, z2, w_c1, b_c1, w_c2, b_c2, out);
}
